// FasterRCNN_2293512536729
// MI455X (gfx1250) — matrix operand order repaired and checked by reading, not compiled, not run on hardware
//
#include <hip/hip_runtime.h>
#include <hip/hip_bf16.h>
#include <stdint.h>

// ---------------------------------------------------------------------------
// Types for CDNA5 WMMA (wave32): v_wmma_f32_16x16x32_bf16
// ---------------------------------------------------------------------------
typedef __bf16 bf16_t;
typedef bf16_t v16bf __attribute__((ext_vector_type(16)));
typedef float  v8f   __attribute__((ext_vector_type(8)));

#define P_TOT   4096          // 64*64 spatial positions
#define C_IN    1024
#define C_MID   512
#define A_TOT   49152         // 4096*12 anchors
#define KEEP_N  2000
#define TOPK_N  300
#define SORT_N  65536
#define XP_W    66            // padded width  (1 + 64 + 1)
#define XP_HW   4356          // 66*66
#define BS_PITCH 144          // LDS row pitch (128 + pad, 288B = 32B aligned)

__device__ __forceinline__ bf16_t f2bf(float x) { return (bf16_t)x; }

// ---------------------------------------------------------------------------
// Kernel 0: pad + convert fx [1024,64,64] f32 -> Xp [1024,66,66] bf16 (zero
// border). Removes all bounds checks / divergence from the GEMM hot loop and
// halves activation read traffic.
// ---------------------------------------------------------------------------
__global__ void pack_x_kernel(const float* __restrict__ fx,
                              bf16_t* __restrict__ Xp) {
  int tid = blockIdx.x * blockDim.x + threadIdx.x;
  if (tid >= C_IN * XP_HW) return;
  int c = tid / XP_HW, r = tid % XP_HW;
  int yy = r / XP_W, xx = r % XP_W;
  float v = 0.0f;
  if (yy >= 1 && yy <= 64 && xx >= 1 && xx <= 64)
    v = fx[(size_t)c * P_TOT + (yy - 1) * 64 + (xx - 1)];
  Xp[tid] = f2bf(v);
}

// ---------------------------------------------------------------------------
// Kernel 1: repack w1 [512,1024,3,3] f32 -> bf16 A-fragments.
// Fragment f = (tap*32 + ocTile)*32 + kChunk ; per-lane 16 bf16 (32B) laid out
// in contiguous k order (same order as the B reads from LDS):
//   lane l : M = l&15 ; K = (l>>4)*16 + j
// ---------------------------------------------------------------------------
__global__ void pack_w_kernel(const float* __restrict__ w1,
                              bf16_t* __restrict__ Apack) {
  int tid = blockIdx.x * blockDim.x + threadIdx.x;
  if (tid >= 9 * 32 * 32 * 32) return;
  int l  = tid & 31;
  int f  = tid >> 5;
  int kc = f & 31;
  int mt = (f >> 5) & 31;
  int t  = f >> 10;
  int m  = l & 15;
  int hi = l >> 4;
  int oc = mt * 16 + m;
  v16bf v;
#pragma unroll
  for (int j = 0; j < 16; ++j) {
    int K  = hi * 16 + j;
    int ic = kc * 32 + K;
    v[j] = f2bf(w1[(size_t)oc * (C_IN * 9) + (size_t)ic * 9 + t]);
  }
  *(v16bf*)(Apack + (size_t)f * 512 + (size_t)l * 16) = v;
}

// ---------------------------------------------------------------------------
// Kernel 2: 3x3 conv + bias + ReLU as bf16 WMMA GEMM.
// Block = 256 thr = 8 waves. Block tile: 128 oc x 32 positions.
// Per stage: 32n x 128k B tile in LDS (8 KB), unconditional coalesced fill,
// then 8 WMMAs per wave (4 K-chunks x 2 N-tiles, A reused across N).
// 72 stages -> 576 WMMAs per wave.
// ---------------------------------------------------------------------------
__global__ void __launch_bounds__(256)
conv3x3_wmma_kernel(const bf16_t* __restrict__ Xp,
                    const bf16_t* __restrict__ Apack,
                    const float* __restrict__ b1,
                    float* __restrict__ Y) {
  __shared__ __align__(32) bf16_t Bs[32][BS_PITCH];   // [n][k]
  const int tid  = threadIdx.x;
  const int lane = tid & 31;
  const int wv   = tid >> 5;                  // wave 0..7
  const int ocGroup = blockIdx.x >> 7;        // 0..3
  const int pTile   = blockIdx.x & 127;       // 0..127
  const int py    = pTile >> 1;               // feature row
  const int xBase = (pTile & 1) << 5;         // 0 or 32
  const int mt    = ocGroup * 8 + wv;         // oc-tile 0..31
  const int n     = lane & 15;
  const int hi    = lane >> 4;
  const int fn    = tid & 31;                 // fill: n
  const int fk    = tid >> 5;                 // fill: k row-group 0..7

  v8f acc0 = {}, acc1 = {};
  for (int t = 0; t < 9; ++t) {
    const int ky = t / 3 - 1, kx = t % 3 - 1;
    const size_t rowBase = (size_t)(py + ky + 1) * XP_W + (xBase + kx + 1) + fn;
    for (int icb = 0; icb < 8; ++icb) {
      // -------- cooperative fill: 32n x 128k, coalesced, unconditional -----
#pragma unroll
      for (int i = 0; i < 16; ++i) {
        int k = i * 8 + fk;
        Bs[fn][k] = Xp[(size_t)(icb * 128 + k) * XP_HW + rowBase];
      }
      __syncthreads();
      // -------- 8 WMMAs: 4 K-chunks x 2 N-tiles ----------------------------
      const size_t fbase = (((size_t)(t * 32 + mt)) * 32 + (size_t)(icb * 4));
      const bf16_t* aptr = Apack + fbase * 512 + (size_t)lane * 16;
      __builtin_prefetch((const void*)(aptr + 4 * 512), 0, 0);  // next stage's A
#pragma unroll
      for (int q = 0; q < 4; ++q) {
        v16bf a  = *(const v16bf*)(aptr + (size_t)q * 512);
        v16bf b0 = *(const v16bf*)(&Bs[n][q * 32 + hi * 16]);
        v16bf b1v = *(const v16bf*)(&Bs[16 + n][q * 32 + hi * 16]);
        acc0 = __builtin_amdgcn_wmma_f32_16x16x32_bf16(
            false, a, false, b0, (short)0, acc0, false, false);
        acc1 = __builtin_amdgcn_wmma_f32_16x16x32_bf16(
            false, a, false, b1v, (short)0, acc1, false, false);
      }
      __syncthreads();
    }
  }
  // write back C (VGPR r -> row r + 8*hi, col n), fused bias + ReLU
  const int pBase = py * 64 + xBase;
#pragma unroll
  for (int r = 0; r < 8; ++r) {
    int oc = mt * 16 + r + 8 * hi;
    float bias = b1[oc];
    Y[(size_t)oc * P_TOT + pBase + n]      = fmaxf(acc0[r] + bias, 0.0f);
    Y[(size_t)oc * P_TOT + pBase + 16 + n] = fmaxf(acc1[r] + bias, 0.0f);
  }
}

// ---------------------------------------------------------------------------
// Kernel 3: both 1x1 convs (12 + 48 output channels), K = 512 dot products.
// ---------------------------------------------------------------------------
__global__ void conv1x1_kernel(const float* __restrict__ Y,
                               const float* __restrict__ w2, const float* __restrict__ b2,
                               const float* __restrict__ w3, const float* __restrict__ b3,
                               float* __restrict__ ox, float* __restrict__ ax) {
  int tid = blockIdx.x * blockDim.x + threadIdx.x;
  if (tid >= 60 * P_TOT) return;
  int p  = tid & 4095;
  int co = tid >> 12;
  const float* w; float bias; float* out;
  if (co < 12) { w = w2 + (size_t)co * 512;        bias = b2[co];      out = ox + (size_t)co * P_TOT; }
  else         { w = w3 + (size_t)(co - 12) * 512; bias = b3[co - 12]; out = ax + (size_t)(co - 12) * P_TOT; }
  float s = bias;
  for (int ic = 0; ic < 512; ++ic)
    s += w[ic] * Y[(size_t)ic * P_TOT + p];
  out[p] = s;
}

// ---------------------------------------------------------------------------
// Kernel 4: anchors + apply_deltas + sigmoid scores.
// Anchor a = p*12 + k ; delta channel = k*4+d ; objectness channel = k.
// ---------------------------------------------------------------------------
__global__ void anchor_kernel(const float* __restrict__ ox,
                              const float* __restrict__ ax,
                              float* __restrict__ scores,
                              float* __restrict__ boxes) {
  int a = blockIdx.x * blockDim.x + threadIdx.x;
  if (a >= A_TOT) return;
  int p = a / 12, k = a % 12;
  int py = p >> 6, px = p & 63;
  const float SC[4] = {32.f, 64.f, 128.f, 256.f};
  const float RT[3] = {0.5f, 1.f, 2.f};
  float s = SC[k / 3], r = RT[k % 3];
  float sq = sqrtf(r);
  float aw = s * sq, ah = s / sq;
  float cx = px * 16.f + 8.f, cy = py * 16.f + 8.f;
  float d0 = ax[(size_t)(k * 4 + 0) * P_TOT + p];
  float d1 = ax[(size_t)(k * 4 + 1) * P_TOT + p];
  float d2 = ax[(size_t)(k * 4 + 2) * P_TOT + p];
  float d3 = ax[(size_t)(k * 4 + 3) * P_TOT + p];
  float xc = cx + d0 * aw, yc = cy + d1 * ah;
  float ww = aw * expf(d2), hh = ah * expf(d3);
  boxes[(size_t)a * 4 + 0] = xc - ww * 0.5f;
  boxes[(size_t)a * 4 + 1] = yc - hh * 0.5f;
  boxes[(size_t)a * 4 + 2] = xc + ww * 0.5f;
  boxes[(size_t)a * 4 + 3] = yc + hh * 0.5f;
  float logit = ox[(size_t)k * P_TOT + p];
  scores[a] = 1.0f / (1.0f + expf(-logit));
}

// ---------------------------------------------------------------------------
// Kernels 5/6: top-2000 via bitonic sort of order-preserving u64 keys
// (descending score, ascending index tie-break — matches jax top_k).
// ---------------------------------------------------------------------------
__global__ void key_init_kernel(const float* __restrict__ scores,
                                unsigned long long* __restrict__ keys) {
  int i = blockIdx.x * blockDim.x + threadIdx.x;
  if (i >= SORT_N) return;
  if (i < A_TOT) {
    unsigned u = __float_as_uint(scores[i]);
    unsigned v = u ^ ((u >> 31) ? 0xFFFFFFFFu : 0x80000000u);  // ascending map
    v = ~v;                                                    // descending
    keys[i] = ((unsigned long long)v << 32) | (unsigned)i;
  } else {
    keys[i] = 0xFFFFFFFFFFFFFFFFull;
  }
}

__global__ void bitonic_step_kernel(unsigned long long* __restrict__ keys,
                                    int j, int k) {
  int i = blockIdx.x * blockDim.x + threadIdx.x;
  int ixj = i ^ j;
  if (ixj > i) {
    unsigned long long a = keys[i], b = keys[ixj];
    bool up = ((i & k) == 0);
    if (up ? (a > b) : (a < b)) { keys[i] = b; keys[ixj] = a; }
  }
}

// ---------------------------------------------------------------------------
// Kernel 7: gather top-2000 boxes/scores, clip to image, validity mask.
// ---------------------------------------------------------------------------
__global__ void extract_kernel(const unsigned long long* __restrict__ keys,
                               const float* __restrict__ scores,
                               const float* __restrict__ boxes,
                               const int* __restrict__ img_h,
                               const int* __restrict__ img_w,
                               float* __restrict__ tb, float* __restrict__ ts,
                               int* __restrict__ tvalid) {
  int i = blockIdx.x * blockDim.x + threadIdx.x;
  if (i >= KEEP_N) return;
  int idx = (int)(keys[i] & 0xFFFFFFFFull);
  float s = scores[idx];
  float W = (float)img_w[0], H = (float)img_h[0];
  float x1 = fminf(fmaxf(boxes[(size_t)idx * 4 + 0], 0.f), W);
  float y1 = fminf(fmaxf(boxes[(size_t)idx * 4 + 1], 0.f), H);
  float x2 = fminf(fmaxf(boxes[(size_t)idx * 4 + 2], 0.f), W);
  float y2 = fminf(fmaxf(boxes[(size_t)idx * 4 + 3], 0.f), H);
  tb[(size_t)i * 4 + 0] = x1; tb[(size_t)i * 4 + 1] = y1;
  tb[(size_t)i * 4 + 2] = x2; tb[(size_t)i * 4 + 3] = y2;
  ts[i] = s;
  tvalid[i] = ((x2 - x1) >= 16.f) && ((y2 - y1) >= 16.f) && (s >= 0.05f);
}

// ---------------------------------------------------------------------------
// Kernel 8: greedy NMS (boxes already sorted by score) + emit top-300 [300,5].
// Single workgroup, everything in LDS (~44 KB of the 320 KB WGP LDS).
// ---------------------------------------------------------------------------
__global__ void __launch_bounds__(1024)
nms_kernel(const float* __restrict__ tb, const float* __restrict__ ts,
           const int* __restrict__ tvalid, float* __restrict__ out) {
  __shared__ float bx[KEEP_N][4];
  __shared__ float ar[KEEP_N];
  __shared__ unsigned char supp[KEEP_N];
  __shared__ unsigned char keep[KEEP_N];
  __shared__ int curKeep;
  int tid = threadIdx.x;
  for (int i = tid; i < KEEP_N; i += 1024) {
    float x1 = tb[(size_t)i * 4 + 0], y1 = tb[(size_t)i * 4 + 1];
    float x2 = tb[(size_t)i * 4 + 2], y2 = tb[(size_t)i * 4 + 3];
    bx[i][0] = x1; bx[i][1] = y1; bx[i][2] = x2; bx[i][3] = y2;
    ar[i] = (x2 - x1) * (y2 - y1);
    supp[i] = tvalid[i] ? 0 : 1;
    keep[i] = 0;
  }
  __syncthreads();
  for (int i = 0; i < KEEP_N; ++i) {
    if (tid == 0) { curKeep = supp[i] ? 0 : 1; keep[i] = (unsigned char)curKeep; }
    __syncthreads();
    if (curKeep) {
      float x1 = bx[i][0], y1 = bx[i][1], x2 = bx[i][2], y2 = bx[i][3];
      float ai = ar[i];
      for (int jn = tid; jn < KEEP_N; jn += 1024) {
        if (!supp[jn]) {
          float lx = fmaxf(x1, bx[jn][0]);
          float ly = fmaxf(y1, bx[jn][1]);
          float rx = fminf(x2, bx[jn][2]);
          float ry = fminf(y2, bx[jn][3]);
          float iw = fmaxf(rx - lx, 0.f), ih = fmaxf(ry - ly, 0.f);
          float inter = iw * ih;
          float iou = inter / fmaxf(ai + ar[jn] - inter, 1e-6f);
          if (iou > 0.7f) supp[jn] = 1;
        }
      }
    }
    __syncthreads();
  }
  if (tid == 0) {
    int o = 0;
    for (int i = 0; i < KEEP_N && o < TOPK_N; ++i) {
      if (keep[i]) {
        out[o * 5 + 0] = bx[i][0]; out[o * 5 + 1] = bx[i][1];
        out[o * 5 + 2] = bx[i][2]; out[o * 5 + 3] = bx[i][3];
        out[o * 5 + 4] = ts[i];
        ++o;
      }
    }
    for (; o < TOPK_N; ++o) {
      out[o * 5 + 0] = 0.f; out[o * 5 + 1] = 0.f; out[o * 5 + 2] = 0.f;
      out[o * 5 + 3] = 0.f; out[o * 5 + 4] = 0.f;
    }
  }
}

// ---------------------------------------------------------------------------
// Host launch
// ---------------------------------------------------------------------------
extern "C" void kernel_launch(void* const* d_in, const int* in_sizes, int n_in,
                              void* d_out, int out_size, void* d_ws, size_t ws_size,
                              hipStream_t stream) {
  (void)in_sizes; (void)n_in; (void)out_size; (void)ws_size;
  const float* fx = (const float*)d_in[0];
  const float* w1 = (const float*)d_in[1];
  const float* b1 = (const float*)d_in[2];
  const float* w2 = (const float*)d_in[3];
  const float* b2 = (const float*)d_in[4];
  const float* w3 = (const float*)d_in[5];
  const float* b3 = (const float*)d_in[6];
  const int* img_h = (const int*)d_in[7];
  const int* img_w = (const int*)d_in[8];
  float* out = (float*)d_out;

  char* ws = (char*)d_ws;
  size_t off = 0;
  auto take = [&](size_t bytes) -> void* {
    void* p = ws + off;
    off += (bytes + 255) & ~(size_t)255;
    return p;
  };
  bf16_t* Xp    = (bf16_t*)take((size_t)C_IN * XP_HW * sizeof(bf16_t));        // 8.9 MB
  bf16_t* Apack = (bf16_t*)take((size_t)9 * 32 * 32 * 512 * sizeof(bf16_t));   // 9.4 MB
  float*  Y     = (float*) take((size_t)C_MID * P_TOT * sizeof(float));        // 8.4 MB
  float*  ox    = (float*) take((size_t)12 * P_TOT * sizeof(float));
  float*  axd   = (float*) take((size_t)48 * P_TOT * sizeof(float));
  float*  sc    = (float*) take((size_t)A_TOT * sizeof(float));
  float*  bxs   = (float*) take((size_t)A_TOT * 4 * sizeof(float));
  unsigned long long* keys = (unsigned long long*)take((size_t)SORT_N * 8);
  float*  tb    = (float*) take((size_t)KEEP_N * 4 * sizeof(float));
  float*  ts    = (float*) take((size_t)KEEP_N * sizeof(float));
  int*    tval  = (int*)   take((size_t)KEEP_N * sizeof(int));

  pack_x_kernel<<<(C_IN * XP_HW + 255) / 256, 256, 0, stream>>>(fx, Xp);
  pack_w_kernel<<<(9 * 32 * 32 * 32) / 256, 256, 0, stream>>>(w1, Apack);
  conv3x3_wmma_kernel<<<512, 256, 0, stream>>>(Xp, Apack, b1, Y);
  conv1x1_kernel<<<(60 * P_TOT) / 256, 256, 0, stream>>>(Y, w2, b2, w3, b3, ox, axd);
  anchor_kernel<<<A_TOT / 256, 256, 0, stream>>>(ox, axd, sc, bxs);
  key_init_kernel<<<SORT_N / 256, 256, 0, stream>>>(sc, keys);
  for (int k = 2; k <= SORT_N; k <<= 1)
    for (int j = k >> 1; j > 0; j >>= 1)
      bitonic_step_kernel<<<SORT_N / 256, 256, 0, stream>>>(keys, j, k);
  extract_kernel<<<(KEEP_N + 255) / 256, 256, 0, stream>>>(keys, sc, bxs, img_h, img_w, tb, ts, tval);
  nms_kernel<<<1, 1024, 0, stream>>>(tb, ts, tval, out);
}
